// NERRelationModel_34299608826143
// MI455X (gfx1250) — matrix operand order repaired and checked by reading, not compiled, not run on hardware
//
#include <hip/hip_runtime.h>
#include <hip/hip_bf16.h>
#include <math.h>
#include <stdint.h>

// ---------------- CDNA5 WMMA types ----------------
typedef __bf16  v16bf __attribute__((ext_vector_type(16)));
typedef __bf16  v8bf  __attribute__((ext_vector_type(8)));
typedef float   v8f   __attribute__((ext_vector_type(8)));

// Problem constants
#define BB   32
#define SS   512
#define HH   768
#define EE   64
#define SPAN 4
#define H1D  4      // gat1 heads
#define D1   128
#define D2   64
#define NREL 6
#define RH   256
#define NERH 256
#define NERL 9

// ======================================================================
// Weight convert+transpose: Bt[n][k] (bf16, n < Npad zero-padded) from
// W[k][n] (f32).  One-time, tiny.
// ======================================================================
__global__ __launch_bounds__(256)
void wconv(const float* __restrict__ W, __bf16* __restrict__ Bt,
           int K, int N, int Npad)
{
    const int idx = blockIdx.x * 256 + threadIdx.x;
    if (idx >= Npad * K) return;
    const int n = idx / K, k = idx % K;
    float v = (n < N) ? W[(size_t)k * N + n] : 0.0f;
    Bt[idx] = (__bf16)v;
}

// ======================================================================
// Generic GEMM: C[M,N] = relu?( A[M,K](f32) * Bt (bf16 [Npad][K]) + bias )
// via v_wmma_f32_16x16x32_bf16.  Block tile 32x256, 8 waves -> 16x64 each.
// Double-buffered LDS; B tiles staged with global_load_async_to_lds_b128
// (ASYNCcnt).  Requires M%32==0, K%32==0, Npad >= gridDim.y*256.
// ======================================================================
#define GP 48   // LDS row pitch in bf16 (96B: keeps 16B alignment for b128)

__global__ __launch_bounds__(256)
void gemm_bf16_wmma(const float* __restrict__ A, const __bf16* __restrict__ Bt,
                    const float* __restrict__ bias, float* __restrict__ C,
                    int M, int N, int K, int relu)
{
    __shared__ __bf16 As[2][32][GP];
    __shared__ __bf16 Bs[2][256][GP];

    const int tid  = threadIdx.x;
    const int lane = tid & 31;
    const int wave = tid >> 5;
    const int wm   = wave >> 2;          // 0..1 : 16-row strip
    const int wn   = wave & 3;           // 0..3 : 64-col strip
    const int half = lane >> 4;          // lane half for frag layout
    const int l16  = lane & 15;

    const int m0 = blockIdx.x * 32;
    const int n0 = blockIdx.y * 256;

    v8f acc[4];
    #pragma unroll
    for (int t = 0; t < 4; ++t) acc[t] = (v8f)0.0f;

    // A staging coords: each thread owns 4 contiguous K of one row
    const int sm = tid >> 3;             // 0..31
    const int sk = (tid & 7) << 2;       // 0,4,...,28

    // ---- async B tile copy: 256 rows x 64B, 4x16B chunks per thread ----
    auto issueB = [&](int buf, int k0) {
        const __bf16* src = Bt + (size_t)(n0 + tid) * K + k0;
        const uint32_t l0 = (uint32_t)(uintptr_t)&Bs[buf][tid][0];
        #pragma unroll
        for (int c = 0; c < 4; ++c) {
            asm volatile("global_load_async_to_lds_b128 %0, %1, off"
                         :: "v"(l0 + 16u * c),
                            "v"((uint64_t)(uintptr_t)(src + 8 * c))
                         : "memory");
        }
    };
    auto gloadA = [&](int k0) -> float4 {
        return *(const float4*)(A + (size_t)(m0 + sm) * K + k0 + sk);
    };
    auto storeA = [&](int buf, float4 av) {
        As[buf][sm][sk + 0] = (__bf16)av.x;
        As[buf][sm][sk + 1] = (__bf16)av.y;
        As[buf][sm][sk + 2] = (__bf16)av.z;
        As[buf][sm][sk + 3] = (__bf16)av.w;
    };

    const int nt = K >> 5;               // #32-wide K tiles

    // prologue: stage tile 0 into buffer 0
    issueB(0, 0);
    float4 av = gloadA(0);
    storeA(0, av);

    for (int i = 0; i < nt; ++i) {
        const int  cur     = i & 1;
        const bool hasNext = (i + 1) < nt;

        if (hasNext) {                   // prefetch next tile into other buffer
            issueB(cur ^ 1, (i + 1) << 5);
            av = gloadA((i + 1) << 5);
            if (i + 2 < nt)
                __builtin_prefetch(A + (size_t)(m0 + sm) * K + ((i + 2) << 5) + sk, 0, 0);
        }

        // wait for CURRENT tile's 4 async copies (per-wave in-order), keep
        // the next tile's 4 in flight.
        if (hasNext) asm volatile("s_wait_asynccnt 0x4" ::: "memory");
        else         asm volatile("s_wait_asynccnt 0x0" ::: "memory");
        __syncthreads();                 // buffer `cur` published

        if (hasNext) storeA(cur ^ 1, av);  // A for next tile (other buffer)

        // ---- A fragment (shared by this wave's 4 tiles) ----
        const int ar  = wm * 16 + l16;
        const int ac0 = half ? 8  : 0;
        const int ac1 = half ? 24 : 16;
        v8bf alo = *(const v8bf*)&As[cur][ar][ac0];
        v8bf ahi = *(const v8bf*)&As[cur][ar][ac1];
        v16bf afrag = __builtin_shufflevector(alo, ahi,
                       0,1,2,3,4,5,6,7,8,9,10,11,12,13,14,15);

        // ---- 4 WMMA tiles along N ----
        const int bc0 = half ? 8  : 0;
        const int bc1 = half ? 24 : 16;
        #pragma unroll
        for (int t = 0; t < 4; ++t) {
            const int bn = wn * 64 + t * 16 + l16;
            v8bf blo = *(const v8bf*)&Bs[cur][bn][bc0];
            v8bf bhi = *(const v8bf*)&Bs[cur][bn][bc1];
            v16bf bfrag = __builtin_shufflevector(blo, bhi,
                           0,1,2,3,4,5,6,7,8,9,10,11,12,13,14,15);
            acc[t] = __builtin_amdgcn_wmma_f32_16x16x32_bf16(
                         false, afrag, false, bfrag, (short)0, acc[t],
                         false, false);
        }
        __syncthreads();                 // done reading `cur` before reuse
    }

    // ---- epilogue: C layout VGPR r -> M = r (+8 for lane>=16), N = l16 ----
    #pragma unroll
    for (int t = 0; t < 4; ++t) {
        const int n = n0 + wn * 64 + t * 16 + l16;
        if (n >= N) continue;
        const float bv = bias ? bias[n] : 0.0f;
        #pragma unroll
        for (int r = 0; r < 8; ++r) {
            const int m = m0 + wm * 16 + r + half * 8;
            float v = acc[t][r] + bv;
            if (relu) v = fmaxf(v, 0.0f);
            C[(size_t)m * N + n] = v;
        }
    }
}

// ======================================================================
// Entity gather + span mean:  x0[b,e,:] = mean(seq[b, st..st+3, :])
// ======================================================================
__global__ __launch_bounds__(256)
void gather_mean(const float* __restrict__ seq, const int* __restrict__ ent,
                 float* __restrict__ x0)
{
    const int blk = blockIdx.x;          // b*64 + e
    const int b = blk >> 6;
    const int st = ent[blk];
    for (int c = threadIdx.x; c < HH; c += 256) {
        float s = 0.0f;
        #pragma unroll
        for (int t = 0; t < SPAN; ++t) {
            int row = st + t; row = row < (SS - 1) ? row : (SS - 1);
            s += seq[((size_t)b * SS + row) * HH + c];
        }
        x0[(size_t)blk * HH + c] = s * 0.25f;
    }
}

// ======================================================================
// NER head 2: out[row,c] = H1[row,:] @ w2[:,c] + b2[c]   (256 -> 9)
// ======================================================================
__global__ __launch_bounds__(256)
void ner_head2(const float* __restrict__ H1, const float* __restrict__ w2,
               const float* __restrict__ b2, float* __restrict__ out)
{
    __shared__ float w2s[NERH * NERL];
    __shared__ float b2s[NERL];
    for (int i = threadIdx.x; i < NERH * NERL; i += 256) w2s[i] = w2[i];
    if (threadIdx.x < NERL) b2s[threadIdx.x] = b2[threadIdx.x];
    __syncthreads();

    const int o = blockIdx.x * 256 + threadIdx.x;
    if (o >= BB * SS * NERL) return;
    const int row = o / NERL;
    const int c   = o % NERL;
    float acc = b2s[c];
    const float* hr = H1 + (size_t)row * NERH;
    #pragma unroll 8
    for (int k = 0; k < NERH; ++k) acc += hr[k] * w2s[k * NERL + c];
    out[o] = acc;
}

// ======================================================================
// Dense GAT attention (one block per (batch, head))
// ======================================================================
__global__ __launch_bounds__(256)
void gat_attn(const float* __restrict__ h, const float* __restrict__ a_src,
              const float* __restrict__ a_dst, const float* __restrict__ bias,
              float* __restrict__ out, int D, int heads)
{
    __shared__ float hs[EE * D1];        // up to 64x128
    __shared__ float alpha[EE * EE];
    __shared__ float sj[EE], di[EE];

    const int b = blockIdx.x, head = blockIdx.y;
    const int tid = threadIdx.x;
    const int W = heads * D;

    for (int e = tid; e < EE * D; e += 256) {
        int j = e / D, d = e % D;
        hs[j * D + d] = h[((size_t)b * EE + j) * W + head * D + d];
    }
    __syncthreads();

    if (tid < EE) {
        float s = 0.0f;
        for (int d = 0; d < D; ++d) s += hs[tid * D + d] * a_src[head * D + d];
        sj[tid] = s;
    } else if (tid < 2 * EE) {
        int i = tid - EE;
        float s = 0.0f;
        for (int d = 0; d < D; ++d) s += hs[i * D + d] * a_dst[head * D + d];
        di[i] = s;
    }
    __syncthreads();

    if (tid < EE) {
        const int i = tid;
        float mx = -1e30f;
        for (int j = 0; j < EE; ++j) {
            if (j == i) continue;
            float l = di[i] + sj[j];
            l = (l >= 0.0f) ? l : 0.2f * l;
            mx = fmaxf(mx, l);
        }
        float sum = 0.0f;
        for (int j = 0; j < EE; ++j) {
            float a = 0.0f;
            if (j != i) {
                float l = di[i] + sj[j];
                l = (l >= 0.0f) ? l : 0.2f * l;
                a = expf(l - mx);
            }
            alpha[i * EE + j] = a;
            sum += a;
        }
        float inv = 1.0f / sum;
        for (int j = 0; j < EE; ++j) alpha[i * EE + j] *= inv;
    }
    __syncthreads();

    for (int e = tid; e < EE * D; e += 256) {
        int i = e / D, d = e % D;
        float acc = 0.0f;
        for (int j = 0; j < EE; ++j) acc += alpha[i * EE + j] * hs[j * D + d];
        out[((size_t)b * EE + i) * W + head * D + d] = acc + bias[head * D + d];
    }
}

// ======================================================================
// LayerNorm + ELU over last dim (one block per row)
// ======================================================================
__global__ __launch_bounds__(256)
void ln_elu(const float* __restrict__ x, const float* __restrict__ g,
            const float* __restrict__ b, float* __restrict__ y, int dim)
{
    __shared__ float red[256];
    const int row = blockIdx.x, tid = threadIdx.x;
    const float* xr = x + (size_t)row * dim;

    float ps = 0.0f;
    for (int k = tid; k < dim; k += 256) ps += xr[k];
    red[tid] = ps; __syncthreads();
    for (int s = 128; s > 0; s >>= 1) { if (tid < s) red[tid] += red[tid + s]; __syncthreads(); }
    const float mean = red[0] / dim; __syncthreads();

    float pv = 0.0f;
    for (int k = tid; k < dim; k += 256) { float d = xr[k] - mean; pv += d * d; }
    red[tid] = pv; __syncthreads();
    for (int s = 128; s > 0; s >>= 1) { if (tid < s) red[tid] += red[tid + s]; __syncthreads(); }
    const float rstd = rsqrtf(red[0] / dim + 1e-5f);

    for (int k = tid; k < dim; k += 256) {
        float v = (xr[k] - mean) * rstd * g[k] + b[k];
        y[(size_t)row * dim + k] = (v > 0.0f) ? v : (expf(v) - 1.0f);
    }
}

// ======================================================================
// Relation combine: one wave per (i,j) pair; LN + relu + dot over RH=256
// ======================================================================
__device__ __forceinline__ float wred(float v) {
    #pragma unroll
    for (int m = 16; m > 0; m >>= 1) v += __shfl_xor(v, m, 32);
    return v;
}

__global__ __launch_bounds__(256)
void rel_combine(const float* __restrict__ P, const float* __restrict__ Q,
                 const float* __restrict__ ln_g, const float* __restrict__ ln_b,
                 const float* __restrict__ w2, const float* __restrict__ b2,
                 float* __restrict__ out)
{
    extern __shared__ float smem[];
    float* Pb = smem;                 // 64*256
    float* Qb = Pb + EE * RH;         // 64*256
    float* gg = Qb + EE * RH;         // 256
    float* bb = gg + RH;              // 256
    float* ww = bb + RH;              // 256

    const int b = blockIdx.x, r = blockIdx.y;
    const int tid = threadIdx.x, lane = tid & 31, wv = tid >> 5;

    const size_t base = ((size_t)r * (BB * EE) + (size_t)b * EE) * RH;
    for (int e = tid; e < EE * RH; e += 256) { Pb[e] = P[base + e]; Qb[e] = Q[base + e]; }
    gg[tid] = ln_g[r * RH + tid];
    bb[tid] = ln_b[r * RH + tid];
    ww[tid] = w2[r * RH + tid];
    __syncthreads();

    const float b2v = b2[r];
    const float invd = 1.0f / (float)RH;

    for (int p = wv; p < EE * EE; p += 8) {
        const int i = p >> 6, j = p & 63;
        float v[8];
        float s = 0.0f;
        #pragma unroll
        for (int e = 0; e < 8; ++e) {
            const int k = lane + 32 * e;
            v[e] = Pb[i * RH + k] + Qb[j * RH + k];
            s += v[e];
        }
        const float mean = wred(s) * invd;
        float vs = 0.0f;
        #pragma unroll
        for (int e = 0; e < 8; ++e) { float d = v[e] - mean; vs += d * d; }
        const float rstd = rsqrtf(wred(vs) * invd + 1e-5f);
        float acc = 0.0f;
        #pragma unroll
        for (int e = 0; e < 8; ++e) {
            const int k = lane + 32 * e;
            float y = (v[e] - mean) * rstd * gg[k] + bb[k];
            y = fmaxf(y, 0.0f);
            acc += y * ww[k];
        }
        acc = wred(acc);
        if (lane == 0)
            out[(((size_t)r * BB + b) * EE + i) * EE + j] = acc + b2v;
    }
}

// ======================================================================
// Host-side launch
// ======================================================================
extern "C" void kernel_launch(void* const* d_in, const int* in_sizes, int n_in,
                              void* d_out, int out_size, void* d_ws, size_t ws_size,
                              hipStream_t stream) {
    const float* seq      = (const float*)d_in[0];
    const int*   ent      = (const int*)  d_in[1];
    const float* ner_w1   = (const float*)d_in[2];
    const float* ner_b1   = (const float*)d_in[3];
    const float* ner_w2   = (const float*)d_in[4];
    const float* ner_b2   = (const float*)d_in[5];
    const float* gat1_w   = (const float*)d_in[6];
    const float* g1_asrc  = (const float*)d_in[7];
    const float* g1_adst  = (const float*)d_in[8];
    const float* gat1_b   = (const float*)d_in[9];
    const float* ln1_g    = (const float*)d_in[10];
    const float* ln1_b    = (const float*)d_in[11];
    const float* gat2_w   = (const float*)d_in[12];
    const float* g2_asrc  = (const float*)d_in[13];
    const float* g2_adst  = (const float*)d_in[14];
    const float* gat2_b   = (const float*)d_in[15];
    const float* ln2_g    = (const float*)d_in[16];
    const float* ln2_b    = (const float*)d_in[17];
    const float* rel_w1   = (const float*)d_in[18];
    const float* rel_b1   = (const float*)d_in[19];
    const float* rel_ln_g = (const float*)d_in[20];
    const float* rel_ln_b = (const float*)d_in[21];
    const float* rel_w2   = (const float*)d_in[22];
    const float* rel_b2   = (const float*)d_in[23];

    float* out = (float*)d_out;
    float* W   = (float*)d_ws;

    // workspace layout (float offsets)
    float* x0  = W;                       // 2048 x 768
    float* Hh  = W + 1572864;             // 16384 x 256 (NER hidden; reused as ao1)
    float* h1  = W + 5767168;             // 2048 x 512
    float* x1  = W + 6815744;             // 2048 x 512
    float* h2  = W + 7864320;             // 2048 x 64
    float* x2  = W + 7995392;             // 2048 x 64
    float* P   = W + 8126464;             // 6 x 2048 x 256
    float* Q   = W + 11272192;            // 6 x 2048 x 256
    __bf16* Wt = (__bf16*)(W + 14417920); // bf16 transposed weights
    float* ao1 = Hh;                      // 2048 x 512 (after ner_head2 done)
    float* ao2 = x0;                      // 2048 x 64  (after g1 done)

    // bf16 weight regions (element offsets into Wt)
    __bf16* ner_w1t = Wt;                 // 256 x 768
    __bf16* gat1_wt = Wt + 196608;        // 512 x 768
    __bf16* gat2_wt = Wt + 589824;        // 256(pad) x 512
    __bf16* relAt   = Wt + 720896;        // 6 x (256 x 64)
    __bf16* relBt   = Wt + 819200;        // 6 x (256 x 64)

    // 0) one-time weight convert + transpose (f32 KxN -> bf16 NpadxK)
    wconv<<<(256 * HH + 255) / 256, 256, 0, stream>>>(ner_w1, ner_w1t, HH, NERH, 256);
    wconv<<<(512 * HH + 255) / 256, 256, 0, stream>>>(gat1_w, gat1_wt, HH, H1D * D1, 512);
    wconv<<<(256 * 512 + 255) / 256, 256, 0, stream>>>(gat2_w, gat2_wt, H1D * D1, D2, 256);
    for (int r = 0; r < NREL; ++r) {
        const float* W1a = rel_w1 + (size_t)r * 2 * D2 * RH;   // rows 0..63
        const float* W1b = W1a + (size_t)D2 * RH;              // rows 64..127
        wconv<<<(256 * D2 + 255) / 256, 256, 0, stream>>>(W1a, relAt + (size_t)r * RH * D2, D2, RH, 256);
        wconv<<<(256 * D2 + 255) / 256, 256, 0, stream>>>(W1b, relBt + (size_t)r * RH * D2, D2, RH, 256);
    }

    // 1) entity gather + span mean
    gather_mean<<<BB * EE, 256, 0, stream>>>(seq, ent, x0);

    // 2) NER layer 1 (WMMA) + relu, then 256->9 head
    gemm_bf16_wmma<<<dim3(512, 1), 256, 0, stream>>>(seq, ner_w1t, ner_b1, Hh,
                                                     BB * SS, NERH, HH, 1);
    ner_head2<<<(BB * SS * NERL + 255) / 256, 256, 0, stream>>>(Hh, ner_w2, ner_b2, out);

    // 3) GAT1: projection (WMMA) + attention + LN/ELU
    gemm_bf16_wmma<<<dim3(64, 2), 256, 0, stream>>>(x0, gat1_wt, nullptr, h1,
                                                    BB * EE, H1D * D1, HH, 0);
    gat_attn<<<dim3(BB, H1D), 256, 0, stream>>>(h1, g1_asrc, g1_adst, gat1_b, ao1, D1, H1D);
    ln_elu<<<BB * EE, 256, 0, stream>>>(ao1, ln1_g, ln1_b, x1, H1D * D1);

    // 4) GAT2: projection (WMMA) + attention + LN/ELU
    gemm_bf16_wmma<<<dim3(64, 1), 256, 0, stream>>>(x1, gat2_wt, nullptr, h2,
                                                    BB * EE, D2, H1D * D1, 0);
    gat_attn<<<dim3(BB, 1), 256, 0, stream>>>(h2, g2_asrc, g2_adst, gat2_b, ao2, D2, 1);
    ln_elu<<<BB * EE, 256, 0, stream>>>(ao2, ln2_g, ln2_b, x2, D2);

    // 5) Relation projections (WMMA): P[r] = x2 @ W1a[r] + b1[r], Q[r] = x2 @ W1b[r]
    for (int r = 0; r < NREL; ++r) {
        gemm_bf16_wmma<<<dim3(64, 1), 256, 0, stream>>>(x2, relAt + (size_t)r * RH * D2,
                                                        rel_b1 + r * RH,
                                                        P + (size_t)r * BB * EE * RH,
                                                        BB * EE, RH, D2, 0);
        gemm_bf16_wmma<<<dim3(64, 1), 256, 0, stream>>>(x2, relBt + (size_t)r * RH * D2,
                                                        nullptr,
                                                        Q + (size_t)r * BB * EE * RH,
                                                        BB * EE, RH, D2, 0);
    }

    // 6) pair combine: LN + relu + dot -> rel_logits
    const size_t shmem = (size_t)(2 * EE * RH + 3 * RH) * sizeof(float);
    rel_combine<<<dim3(BB, NREL), 256, shmem, stream>>>(P, Q, rel_ln_g, rel_ln_b,
                                                        rel_w2, rel_b2,
                                                        out + BB * SS * NERL);
}
